// BERT_58677843198446
// MI455X (gfx1250) — matrix operand order repaired and checked by reading, not compiled, not run on hardware
//
#include <hip/hip_runtime.h>

typedef __attribute__((ext_vector_type(16))) __bf16 v16bf;
typedef __attribute__((ext_vector_type(8)))  float  v8f;

#define NN    20000
#define EE    320000
#define INDIM 256
#define DD    512
#define FFD   2048
#define OUTD  128
#define C3    1536
#define HEADS 128

// ---------------- elementwise / conversion kernels ----------------

__global__ void k_fill_f32(float* __restrict__ p, float v, int n) {
  int i = blockIdx.x * 256 + threadIdx.x;
  if (i < n) p[i] = v;
}

__global__ void k_cvt_bf16(const float* __restrict__ in, __bf16* __restrict__ out, int n) {
  int i = blockIdx.x * 256 + threadIdx.x;
  if (i < n) out[i] = (__bf16)in[i];
}

__global__ void k_relu_cvt_bf16(const float* __restrict__ in, __bf16* __restrict__ out, int n) {
  int i = blockIdx.x * 256 + threadIdx.x;
  if (i < n) out[i] = (__bf16)fmaxf(in[i], 0.f);
}

// W: K x N row-major (f32)  ->  Wt: N x K row-major (bf16)
__global__ void k_transpose_cvt(const float* __restrict__ W, __bf16* __restrict__ Wt,
                                int K, int N) {
  int i = blockIdx.x * 256 + threadIdx.x;
  if (i < K * N) {
    int n = i / K, k = i - n * K;
    Wt[i] = (__bf16)W[(size_t)k * N + n];
  }
}

// ---------------- bf16 WMMA GEMM ----------------
// C[M,N] = A[M,K] * Bt[N,K]^T  (+bias) (+addend) (opt relu) -> f32 and/or bf16
// Block tile 128x128, 8 waves (2 cols x 4 rows of 32x64 wave tiles), K-step 32.
__global__ __launch_bounds__(256, 2)
void k_gemm_bf16(const __bf16* __restrict__ A, const __bf16* __restrict__ Bt,
                 const float* __restrict__ bias, const float* __restrict__ addend,
                 float* __restrict__ Cf, __bf16* __restrict__ Cb,
                 int M, int N, int K, int relu)
{
  constexpr int LDA = 40;                  // 32 + 8 bf16 pad; 80B rows stay 16B-aligned
  __shared__ __bf16 As[128 * LDA];
  __shared__ __bf16 Bs[128 * LDA];

  const int tid   = threadIdx.x;
  const int lane  = tid & 31;
  const int wave  = tid >> 5;
  const int wm    = (wave >> 1) * 32;      // wave row offset within 128
  const int wn    = (wave & 1) * 64;       // wave col offset within 128
  const int m0    = blockIdx.y * 128;
  const int n0    = blockIdx.x * 128;
  const int mlane = lane & 15;
  const int kh    = lane >> 4;             // lane half (wave32)

  v8f acc[2][4] = {};

  const int lrow = tid >> 1;               // 0..127 : tile row this thread stages
  const int lcol = (tid & 1) * 16;         // 0 or 16 : k-segment

  union Frag { uint4 u[2]; v16bf v; };

  for (int k0 = 0; k0 < K; k0 += 32) {
    // stage A tile (128x32 bf16)
    uint4 a0 = {0,0,0,0}, a1 = {0,0,0,0};
    int gm = m0 + lrow;
    if (gm < M) {
      const uint4* s = (const uint4*)(A + (size_t)gm * K + k0 + lcol);
      a0 = s[0]; a1 = s[1];
    }
    *(uint4*)(&As[lrow * LDA + lcol])     = a0;
    *(uint4*)(&As[lrow * LDA + lcol + 8]) = a1;
    // stage B tile (128 n-rows x 32 k, from pre-transposed Bt)
    uint4 b0 = {0,0,0,0}, b1 = {0,0,0,0};
    int gn = n0 + lrow;
    if (gn < N) {
      const uint4* s = (const uint4*)(Bt + (size_t)gn * K + k0 + lcol);
      b0 = s[0]; b1 = s[1];
    }
    *(uint4*)(&Bs[lrow * LDA + lcol])     = b0;
    *(uint4*)(&Bs[lrow * LDA + lcol + 8]) = b1;

    __syncthreads();

    Frag afr[2], bfr[4];
    // A fragment: lane(m=mlane,kh): elems 0..7 = K[kh*8..+7], elems 8..15 = K[16+kh*8..+7]
#pragma unroll
    for (int mt = 0; mt < 2; ++mt) {
      const __bf16* p = &As[(wm + mt * 16 + mlane) * LDA + kh * 8];
      afr[mt].u[0] = *(const uint4*)(p);
      afr[mt].u[1] = *(const uint4*)(p + 16);
    }
    // B fragment: lane(n=mlane,kh): elems 0..7 = K[kh*8..+7], elems 8..15 = K[16+kh*8..+7]
#pragma unroll
    for (int nt = 0; nt < 4; ++nt) {
      const __bf16* p = &Bs[(wn + nt * 16 + mlane) * LDA + kh * 8];
      bfr[nt].u[0] = *(const uint4*)(p);
      bfr[nt].u[1] = *(const uint4*)(p + 16);
    }
#pragma unroll
    for (int mt = 0; mt < 2; ++mt)
#pragma unroll
      for (int nt = 0; nt < 4; ++nt)
        acc[mt][nt] = __builtin_amdgcn_wmma_f32_16x16x32_bf16(
            false, afr[mt].v, false, bfr[nt].v, (short)0, acc[mt][nt], false, false);

    __syncthreads();
  }

  // epilogue: C/D layout -> lane col = mlane, VGPR r -> row r + 8*kh
#pragma unroll
  for (int mt = 0; mt < 2; ++mt) {
    int gmBase = m0 + wm + mt * 16 + kh * 8;
#pragma unroll
    for (int nt = 0; nt < 4; ++nt) {
      int gn = n0 + wn + nt * 16 + mlane;
      if (gn >= N) continue;
      float bv = bias ? bias[gn] : 0.f;
#pragma unroll
      for (int r = 0; r < 8; ++r) {
        int gm = gmBase + r;
        if (gm < M) {
          float v = acc[mt][nt][r] + bv;
          size_t o = (size_t)gm * N + gn;
          if (addend) v += addend[o];
          if (relu)   v = fmaxf(v, 0.f);
          if (Cf) Cf[o] = v;
          if (Cb) Cb[o] = (__bf16)v;
        }
      }
    }
  }
}

// ---------------- depthwise conv over node dim (K=5, SAME) ----------------
__global__ void k_dwconv(const float* __restrict__ qkv, const float* __restrict__ w,
                         float* __restrict__ ms) {
  int i = blockIdx.x * 256 + threadIdx.x;
  if (i >= NN * C3) return;
  int n = i / C3, c = i - n * C3;
  const float* wc = w + c * 5;
  float acc = 0.f;
#pragma unroll
  for (int j = 0; j < 5; ++j) {
    int nn = n + j - 2;
    if (nn >= 0 && nn < NN) acc += qkv[(size_t)nn * C3 + c] * wc[j];
  }
  ms[i] = acc;
}

// ---------------- lite-MLA: kv = sum_n relu(k) (x) [v,1] ----------------
__global__ __launch_bounds__(256)
void k_kv(const float* __restrict__ qkv, const float* __restrict__ ms,
          float* __restrict__ kv) {
  int h = blockIdx.x;
  const float* src = (h < 64) ? qkv : ms;
  int base = (h & 63) * 24;
  int start = blockIdx.y * 2500;
  int end = start + 2500; if (end > NN) end = NN;
  float a[72];
#pragma unroll
  for (int i = 0; i < 72; ++i) a[i] = 0.f;
  for (int n = start + threadIdx.x; n < end; n += 256) {
    const float* p = src + (size_t)n * C3 + base;
    float kd[8], ve[9];
#pragma unroll
    for (int d = 0; d < 8; ++d) kd[d] = fmaxf(p[8 + d], 0.f);
#pragma unroll
    for (int e = 0; e < 8; ++e) ve[e] = p[16 + e];
    ve[8] = 1.f;
#pragma unroll
    for (int d = 0; d < 8; ++d)
#pragma unroll
      for (int e = 0; e < 9; ++e) a[d * 9 + e] += kd[d] * ve[e];
  }
  __shared__ float s[72];
  if (threadIdx.x < 72) s[threadIdx.x] = 0.f;
  __syncthreads();
#pragma unroll
  for (int i = 0; i < 72; ++i) atomicAdd(&s[i], a[i]);
  __syncthreads();
  if (threadIdx.x < 72) atomicAdd(&kv[h * 72 + threadIdx.x], s[threadIdx.x]);
}

// ---------------- lite-MLA: o = (relu(q) @ kv) num/den ----------------
__global__ __launch_bounds__(256)
void k_attn_out(const float* __restrict__ qkv, const float* __restrict__ ms,
                const float* __restrict__ kv, __bf16* __restrict__ ob) {
  __shared__ float skv[HEADS * 72];
  for (int i = threadIdx.x; i < HEADS * 72; i += 256) skv[i] = kv[i];
  __syncthreads();
  int idx = blockIdx.x * 256 + threadIdx.x;
  if (idx >= NN * HEADS) return;
  int h = idx & (HEADS - 1);
  int n = idx >> 7;
  const float* src = (h < 64) ? qkv : ms;
  const float* p = src + (size_t)n * C3 + (h & 63) * 24;
  float q[8];
#pragma unroll
  for (int d = 0; d < 8; ++d) q[d] = fmaxf(p[d], 0.f);
  float num[9];
#pragma unroll
  for (int e = 0; e < 9; ++e) num[e] = 0.f;
  const float* khv = &skv[h * 72];
#pragma unroll
  for (int d = 0; d < 8; ++d)
#pragma unroll
    for (int e = 0; e < 9; ++e) num[e] += q[d] * khv[d * 9 + e];
  float inv = 1.f / (num[8] + 1e-15f);
  size_t o = (size_t)n * 1024 + (size_t)h * 8;
#pragma unroll
  for (int e = 0; e < 8; ++e) ob[o + e] = (__bf16)(num[e] * inv);
}

// ---------------- out = base + LN(x)*g + beta  (D=512, block per row) ----------------
__global__ __launch_bounds__(256)
void k_ln_add(const float* __restrict__ base, const float* __restrict__ x,
              const float* __restrict__ g, const float* __restrict__ beta,
              float* __restrict__ outf, __bf16* __restrict__ outb) {
  __shared__ float s1[256], s2[256];
  int row = blockIdx.x, t = threadIdx.x;
  const float* xr = x + (size_t)row * DD;
  float x0 = xr[t], x1 = xr[t + 256];
  s1[t] = x0 + x1;
  s2[t] = x0 * x0 + x1 * x1;
  __syncthreads();
  for (int o = 128; o > 0; o >>= 1) {
    if (t < o) { s1[t] += s1[t + o]; s2[t] += s2[t + o]; }
    __syncthreads();
  }
  float mean = s1[0] * (1.f / DD);
  float var  = s2[0] * (1.f / DD) - mean * mean;
  float inv  = rsqrtf(var + 1e-5f);
  const float* br = base + (size_t)row * DD;
  float y0 = br[t]       + (x0 - mean) * inv * g[t]       + beta[t];
  float y1 = br[t + 256] + (x1 - mean) * inv * g[t + 256] + beta[t + 256];
  float* of = outf + (size_t)row * DD;
  of[t] = y0; of[t + 256] = y1;
  if (outb) {
    __bf16* ob = outb + (size_t)row * DD;
    ob[t] = (__bf16)y0; ob[t + 256] = (__bf16)y1;
  }
}

// ---------------- graph normalization + scatter ----------------
__global__ void k_deg_edges(const int* __restrict__ ei, const float* __restrict__ attr,
                            float* __restrict__ deg) {
  int e = blockIdx.x * 256 + threadIdx.x;
  if (e < EE) atomicAdd(&deg[ei[e]], attr[e]);
}

__global__ void k_dis(const float* __restrict__ deg, float* __restrict__ dis) {
  int i = blockIdx.x * 256 + threadIdx.x;
  if (i < NN) {
    float d = deg[i];
    dis[i] = d > 0.f ? rsqrtf(fmaxf(d, 1e-12f)) : 0.f;
  }
}

__global__ void k_agg_self(const float* __restrict__ src, const float* __restrict__ dis,
                           float* __restrict__ agg) {
  int i = blockIdx.x * 256 + threadIdx.x;
  if (i < NN * DD) {
    int n = i >> 9;
    float s = dis[n];
    agg[i] = s * s * src[i];
  }
}

__global__ __launch_bounds__(256)
void k_agg_edges(const int* __restrict__ ei, const float* __restrict__ attr,
                 const float* __restrict__ dis, const float* __restrict__ src,
                 float* __restrict__ agg) {
  int e = blockIdx.x;
  int r = ei[e], c = ei[EE + e];
  float nw = dis[r] * attr[e] * dis[c];
  const float* sc = src + (size_t)c * DD;
  float* ar = agg + (size_t)r * DD;
  for (int t = threadIdx.x; t < DD; t += 256)
    atomicAdd(&ar[t], nw * sc[t]);
}

// ---------------- host orchestration ----------------
static inline int cdiv(int a, int b) { return (a + b - 1) / b; }

static void launch_gemm(const __bf16* A, const __bf16* Bt, const float* bias,
                        const float* addend, float* Cf, __bf16* Cb,
                        int M, int N, int K, int relu, hipStream_t s) {
  dim3 grid(cdiv(N, 128), cdiv(M, 128));
  k_gemm_bf16<<<grid, dim3(256), 0, s>>>(A, Bt, bias, addend, Cf, Cb, M, N, K, relu);
}

extern "C" void kernel_launch(void* const* d_in, const int* in_sizes, int n_in,
                              void* d_out, int out_size, void* d_ws, size_t ws_size,
                              hipStream_t stream) {
  const float* x      = (const float*)d_in[0];
  const int*   ei     = (const int*)  d_in[1];
  const float* eattr  = (const float*)d_in[2];
  const float* pe     = (const float*)d_in[4];
  const float* W_in   = (const float*)d_in[5];
  const float* b_in   = (const float*)d_in[6];
  const float* W_qkv  = (const float*)d_in[7];
  const float* conv_w = (const float*)d_in[8];
  const float* W_proj = (const float*)d_in[9];
  const float* b_proj = (const float*)d_in[10];
  const float* g1     = (const float*)d_in[11];
  const float* beta1  = (const float*)d_in[12];
  const float* W1     = (const float*)d_in[13];
  const float* bf1    = (const float*)d_in[14];
  const float* W2     = (const float*)d_in[15];
  const float* bf2    = (const float*)d_in[16];
  const float* g2     = (const float*)d_in[17];
  const float* beta2  = (const float*)d_in[18];
  const float* W_out  = (const float*)d_in[19];
  const float* b_out  = (const float*)d_in[20];
  float* out = (float*)d_out;

  char* ws = (char*)d_ws;
  size_t off = 0;
  auto take = [&](size_t bytes) { char* p = ws + off; off += (bytes + 255) & ~(size_t)255; return p; };

  __bf16* wt_in   = (__bf16*)take((size_t)DD * INDIM * 2);   // [DD][INDIM]
  __bf16* wt_qkv  = (__bf16*)take((size_t)C3 * DD * 2);      // [C3][DD]
  __bf16* wt_proj = (__bf16*)take((size_t)DD * 1024 * 2);    // [DD][1024]
  __bf16* wt_1    = (__bf16*)take((size_t)FFD * DD * 2);     // [FFD][DD]
  __bf16* wt_2    = (__bf16*)take((size_t)DD * FFD * 2);     // [DD][FFD]
  __bf16* wt_out  = (__bf16*)take((size_t)OUTD * DD * 2);    // [OUTD][DD]
  __bf16* x_b     = (__bf16*)take((size_t)NN * INDIM * 2);
  float*  h_f     = (float*) take((size_t)NN * DD * 4);
  __bf16* h_b     = (__bf16*)take((size_t)NN * DD * 2);
  float*  kv      = (float*) take((size_t)HEADS * 72 * 4);
  __bf16* o_b     = (__bf16*)take((size_t)NN * 1024 * 2);
  float*  src_f   = (float*) take((size_t)NN * DD * 4);
  float*  deg     = (float*) take((size_t)NN * 4);
  float*  dis     = (float*) take((size_t)NN * 4);

  // phase-reused big region: attention phase uses qkv/ms; post-attention temps overlap them
  char* R = ws + off;
  float*  qkv_f  = (float*) R;                                    // [NN][C3]
  float*  ms_f   = (float*)(R + (size_t)NN * C3 * 4);             // [NN][C3]
  float*  src2_f = (float*) R;                                    // [NN][DD]   (after attn)
  __bf16* src_b  = (__bf16*)(R + (size_t)NN * DD * 4);            // [NN][DD]
  __bf16* ffh_b  = (__bf16*)((char*)src_b + (size_t)NN * DD * 2); // [NN][FFD]
  float*  ff_f   = (float*) ((char*)ffh_b + (size_t)NN * FFD * 2);// [NN][DD]
  float*  agg_f  = (float*) ((char*)ff_f + (size_t)NN * DD * 4);  // [NN][DD]
  __bf16* agg_b  = (__bf16*)((char*)agg_f + (size_t)NN * DD * 4); // [NN][DD]

  // ---- weight prep (bf16, transposed to N x K) ----
  k_transpose_cvt<<<cdiv(INDIM * DD, 256), 256, 0, stream>>>(W_in,   wt_in,   INDIM, DD);
  k_transpose_cvt<<<cdiv(DD * C3, 256),    256, 0, stream>>>(W_qkv,  wt_qkv,  DD,    C3);
  k_transpose_cvt<<<cdiv(1024 * DD, 256),  256, 0, stream>>>(W_proj, wt_proj, 1024,  DD);
  k_transpose_cvt<<<cdiv(DD * FFD, 256),   256, 0, stream>>>(W1,     wt_1,    DD,    FFD);
  k_transpose_cvt<<<cdiv(FFD * DD, 256),   256, 0, stream>>>(W2,     wt_2,    FFD,   DD);
  k_transpose_cvt<<<cdiv(DD * OUTD, 256),  256, 0, stream>>>(W_out,  wt_out,  DD,    OUTD);
  k_cvt_bf16<<<cdiv(NN * INDIM, 256), 256, 0, stream>>>(x, x_b, NN * INDIM);

  // ---- h = x@W_in + b_in + pe ----
  launch_gemm(x_b, wt_in, b_in, pe, h_f, h_b, NN, DD, INDIM, 0, stream);

  // ---- qkv = h@W_qkv ; ms = depthwise conv ----
  launch_gemm(h_b, wt_qkv, nullptr, nullptr, qkv_f, nullptr, NN, C3, DD, 0, stream);
  k_dwconv<<<cdiv(NN * C3, 256), 256, 0, stream>>>(qkv_f, conv_w, ms_f);

  // ---- lite-MLA ----
  k_fill_f32<<<cdiv(HEADS * 72, 256), 256, 0, stream>>>(kv, 0.f, HEADS * 72);
  k_kv<<<dim3(HEADS, 8), 256, 0, stream>>>(qkv_f, ms_f, kv);
  k_attn_out<<<cdiv(NN * HEADS, 256), 256, 0, stream>>>(qkv_f, ms_f, kv, o_b);

  // ---- src2 = o@W_proj + b_proj ; src = h + LN(src2) ----
  launch_gemm(o_b, wt_proj, b_proj, nullptr, src2_f, nullptr, NN, DD, 1024, 0, stream);
  k_ln_add<<<NN, 256, 0, stream>>>(h_f, src2_f, g1, beta1, src_f, src_b);

  // ---- FF: ff = relu(src@W1+bf1)@W2 + bf2 ; src += LN(ff) ----
  launch_gemm(src_b, wt_1, bf1, nullptr, nullptr, ffh_b, NN, FFD, DD, 1, stream);
  launch_gemm(ffh_b, wt_2, bf2, nullptr, ff_f, nullptr, NN, DD, FFD, 0, stream);
  k_ln_add<<<NN, 256, 0, stream>>>(src_f, ff_f, g2, beta2, src_f, (__bf16*)nullptr);

  // ---- graph norm + scatter ----
  k_fill_f32<<<cdiv(NN, 256), 256, 0, stream>>>(deg, 1.f, NN);     // self-loop weight
  k_deg_edges<<<cdiv(EE, 256), 256, 0, stream>>>(ei, eattr, deg);
  k_dis<<<cdiv(NN, 256), 256, 0, stream>>>(deg, dis);
  k_agg_self<<<cdiv(NN * DD, 256), 256, 0, stream>>>(src_f, dis, agg_f);
  k_agg_edges<<<EE, 256, 0, stream>>>(ei, eattr, dis, src_f, agg_f);

  // ---- out = relu(agg)@W_out + b_out ----
  k_relu_cvt_bf16<<<cdiv(NN * DD, 256), 256, 0, stream>>>(agg_f, agg_b, NN * DD);
  launch_gemm(agg_b, wt_out, b_out, nullptr, out, nullptr, NN, OUTD, DD, 0, stream);
}
